// TransformerBlock_72541997629782
// MI455X (gfx1250) — matrix operand order repaired and checked by reading, not compiled, not run on hardware
//
#include <hip/hip_runtime.h>
#include <hip/hip_bf16.h>
#include <math.h>
#include <stdint.h>

#define BATCH 2
#define SEQ   2048
#define EMB   1024
#define HEADS 16
#define HDIM  64
#define FFDIM 4096
#define MROWS (BATCH*SEQ)   // 4096

typedef _Float16 half_t;
typedef __attribute__((ext_vector_type(16))) _Float16 v16h;
typedef __attribute__((ext_vector_type(8)))  _Float16 v8h;
typedef __attribute__((ext_vector_type(8)))  float    v8f;

// ---------------------------------------------------------------------------
// WMMA helpers (CDNA5 16x16x32 f16 -> f32)
// ---------------------------------------------------------------------------
__device__ __forceinline__ v8f wmma_fp16(v16h a, v16h b, v8f c) {
  return __builtin_amdgcn_wmma_f32_16x16x32_f16(false, a, false, b, (short)0, c,
                                                false, false);
}

// A-matrix 16x32 f16 fragment (row-major source, ld in elements).
// ISA layout: lanes 0-15 -> rows M=0..15, K chunks {0..7,16..23};
//             lanes 16-31 -> same rows, K chunks {8..15,24..31}.
__device__ __forceinline__ v16h load_a_frag(const half_t* p, int ld, int m0,
                                            int k0, int lane) {
  int row = m0 + (lane & 15);
  int kk  = k0 + ((lane & 16) ? 8 : 0);
  const half_t* r = p + (size_t)row * ld + kk;
  v8h lo = *(const v8h*)(r);
  v8h hi = *(const v8h*)(r + 16);
  v16h a;
#pragma unroll
  for (int i = 0; i < 8; ++i) { a[i] = lo[i]; a[i + 8] = hi[i]; }
  return a;
}

// B-matrix 32x16 f16 fragment, sourced from a transposed [N,K] row-major array
// so each lane's 16 K-values are contiguous.
// ISA layout: lanes 0-15 -> K chunks {0..7,16..23} (N=lane), lanes 16-31 -> K chunks {8..15,24..31}.
__device__ __forceinline__ v16h load_b_frag(const half_t* p, int ld, int n0,
                                            int k0, int lane) {
  int n  = n0 + (lane & 15);
  int kk = k0 + ((lane & 16) ? 8 : 0);
  const half_t* r = p + (size_t)n * ld + kk;
  v8h lo = *(const v8h*)(r);
  v8h hi = *(const v8h*)(r + 16);
  v16h b;
#pragma unroll
  for (int i = 0; i < 8; ++i) { b[i] = lo[i]; b[i + 8] = hi[i]; }
  return b;
}

// ---------------------------------------------------------------------------
// CDNA5 async-to-LDS copy (ASYNCcnt path) — bypasses VGPRs entirely.
// ---------------------------------------------------------------------------
__device__ __forceinline__ void async_ld_b128(uint32_t lds_off, const void* g) {
  asm volatile("global_load_async_to_lds_b128 %0, %1, off"
               :: "v"(lds_off), "v"(g) : "memory");
}
__device__ __forceinline__ void wait_async_le4() {
  asm volatile("s_wait_asynccnt 0x4" ::: "memory");
}
__device__ __forceinline__ void wait_async_0() {
  asm volatile("s_wait_asynccnt 0x0" ::: "memory");
}

// ---------------------------------------------------------------------------
// Weight prep: fp32 [K,N] -> f16 [N,K] (transposed for contiguous B-frags)
// ---------------------------------------------------------------------------
__global__ __launch_bounds__(256)
void convert_transpose_f16(const float* __restrict__ W, half_t* __restrict__ WT,
                           int K, int N) {
  int idx = blockIdx.x * 256 + threadIdx.x;
  if (idx >= K * N) return;
  int k = idx / N, n = idx % N;
  WT[(size_t)n * K + k] = (half_t)W[idx];
}

// ---------------------------------------------------------------------------
// LayerNorm (fp32 in) -> f16 out, one 256-thread block per row of EMB=1024
// ---------------------------------------------------------------------------
__global__ __launch_bounds__(256)
void layernorm_to_f16(const float* __restrict__ x, const float* __restrict__ g,
                      const float* __restrict__ s, half_t* __restrict__ out) {
  int row = blockIdx.x;
  const float* xr = x + (size_t)row * EMB;
  float vals[4];
  float sum = 0.f, sq = 0.f;
#pragma unroll
  for (int i = 0; i < 4; ++i) {
    float v = xr[threadIdx.x + i * 256];
    vals[i] = v; sum += v; sq += v * v;
  }
#pragma unroll
  for (int off = 16; off >= 1; off >>= 1) {
    sum += __shfl_xor(sum, off, 32);
    sq  += __shfl_xor(sq,  off, 32);
  }
  __shared__ float s1[8], s2[8];
  int wave = threadIdx.x >> 5, lane = threadIdx.x & 31;
  if (lane == 0) { s1[wave] = sum; s2[wave] = sq; }
  __syncthreads();
  float S = 0.f, S2 = 0.f;
#pragma unroll
  for (int w = 0; w < 8; ++w) { S += s1[w]; S2 += s2[w]; }
  float mean = S * (1.0f / EMB);
  float var  = S2 * (1.0f / EMB) - mean * mean;
  float rstd = rsqrtf(var + 1e-5f);
#pragma unroll
  for (int i = 0; i < 4; ++i) {
    int c = threadIdx.x + i * 256;
    out[(size_t)row * EMB + c] = (half_t)(g[c] * (vals[i] - mean) * rstd + s[c]);
  }
}

// ---------------------------------------------------------------------------
// WMMA GEMM: C[M,N] = A[M,K](f16) * BT[N,K](f16)^T  with fused epilogues.
// 256 threads = 8 waves; block tile 128x128; wave tile 32x64 (2x4 frags,
// 8 WMMA per 32-K step). Double-buffered async-to-LDS staging (ASYNCcnt),
// last K-step peeled so the steady-state loop is branch-free.
// ---------------------------------------------------------------------------
#define BM 128
#define BN 128
#define BK 32
#define LDS_STRIDE 40   // halves; row pitch 80B = multiple of 16B

enum { EPI_QK = 0, EPI_VT = 1, EPI_GELU = 2, EPI_RES32 = 3 };

__global__ __launch_bounds__(256)
void gemm_wmma(const half_t* __restrict__ A, const half_t* __restrict__ BT,
               const float* __restrict__ bias, const float* __restrict__ resid,
               void* __restrict__ out, int Mdim, int Ndim, int Kdim, int epi) {
  __shared__ __align__(16) half_t sA[2][BM * LDS_STRIDE];
  __shared__ __align__(16) half_t sB[2][BN * LDS_STRIDE];

  int tid  = threadIdx.x;
  int lane = tid & 31;
  int wave = tid >> 5;
  int wm = wave & 3;       // 4 waves down M  (32 rows each)
  int wn = wave >> 2;      // 2 waves across N (64 cols each)
  int m_blk = blockIdx.y * BM;
  int n_blk = blockIdx.x * BN;

  // staging assignment: thread t -> row (t>>1), 32-byte chunk (t&1)
  int r  = tid >> 1;
  int ck = (tid & 1) * 16;          // halves

  v8f acc[2][4] = {};

  auto issue = [&](int buf, int k0) {
    const half_t* ga = A  + (size_t)(m_blk + r) * Kdim + k0 + ck;
    const half_t* gb = BT + (size_t)(n_blk + r) * Kdim + k0 + ck;
    uint32_t la = (uint32_t)(uintptr_t)&sA[buf][r * LDS_STRIDE + ck];
    uint32_t lb = (uint32_t)(uintptr_t)&sB[buf][r * LDS_STRIDE + ck];
    async_ld_b128(la,      ga);
    async_ld_b128(la + 16, ga + 8);
    async_ld_b128(lb,      gb);
    async_ld_b128(lb + 16, gb + 8);
  };

  auto compute = [&](int buf) {
    const half_t* cA = sA[buf];
    const half_t* cB = sB[buf];
    v16h a0 = load_a_frag(cA, LDS_STRIDE, wm * 32,      0, lane);
    v16h a1 = load_a_frag(cA, LDS_STRIDE, wm * 32 + 16, 0, lane);
    v16h b[4];
#pragma unroll
    for (int j = 0; j < 4; ++j)
      b[j] = load_b_frag(cB, LDS_STRIDE, wn * 64 + j * 16, 0, lane);
#pragma unroll
    for (int j = 0; j < 4; ++j) {
      acc[0][j] = wmma_fp16(a0, b[j], acc[0][j]);
      acc[1][j] = wmma_fp16(a1, b[j], acc[1][j]);
    }
  };

  issue(0, 0);
  int nsteps = Kdim / BK;
  for (int st = 0; st < nsteps - 1; ++st) {
    int cur = st & 1;
    issue(cur ^ 1, (st + 1) * BK);   // prefetch next tile into other buffer
    wait_async_le4();                // newest 4 outstanding; older batch done
    __syncthreads();
    compute(cur);
    __syncthreads();                 // protect buffer reuse for next issue
  }
  wait_async_0();
  __syncthreads();
  compute((nsteps - 1) & 1);

  // Epilogue. C/D layout: lane holds col n = lane&15, rows m = v + 8*(lane>=16)
  int hi = (lane >> 4) & 1;
  int ln = lane & 15;
#pragma unroll
  for (int i = 0; i < 2; ++i) {
#pragma unroll
    for (int j = 0; j < 4; ++j) {
#pragma unroll
      for (int v = 0; v < 8; ++v) {
        int m = m_blk + wm * 32 + i * 16 + v + hi * 8;
        int n = n_blk + wn * 64 + j * 16 + ln;
        float val = acc[i][j][v];
        if (epi == EPI_QK) {
          int b = m / SEQ, s = m % SEQ, h = n >> 6, d = n & 63;
          ((half_t*)out)[(((size_t)(b * HEADS + h) * SEQ) + s) * HDIM + d] = (half_t)val;
        } else if (epi == EPI_VT) {
          int b = m / SEQ, s = m % SEQ, h = n >> 6, d = n & 63;
          ((half_t*)out)[(((size_t)(b * HEADS + h) * HDIM) + d) * SEQ + s] = (half_t)val;
        } else if (epi == EPI_GELU) {
          val += bias[n];
          float t = 0.7978845608028654f * (val + 0.044715f * val * val * val);
          float gel = 0.5f * val * (1.0f + tanhf(t));
          ((half_t*)out)[(size_t)m * Ndim + n] = (half_t)gel;
        } else { // EPI_RES32
          val += bias[n] + resid[(size_t)m * Ndim + n];
          ((float*)out)[(size_t)m * Ndim + n] = val;
        }
      }
    }
  }
}

// ---------------------------------------------------------------------------
// Flash attention: one wave per 32-query tile (two 16-row fragments so K/V
// B-fragments are shared), online softmax over 32-key steps.
// Q,K f16 [b,h,s,d]; V f16 transposed [b,h,d,s]; scale = 1/sqrt(64).
// 16 WMMA per 32-key step (8 for S = Q*K^T, 8 for ctx += P*V).
// ---------------------------------------------------------------------------
__global__ __launch_bounds__(32)
void attn_flash_wmma(const half_t* __restrict__ q, const half_t* __restrict__ k,
                     const half_t* __restrict__ vT, half_t* __restrict__ ctx) {
  __shared__ __align__(16) half_t sP[32 * LDS_STRIDE];

  int lane = threadIdx.x & 31;
  int hi = (lane >> 4) & 1;
  int ln = lane & 15;
  int bh = blockIdx.y;                 // 0..BATCH*HEADS-1
  int b = bh / HEADS, h = bh % HEADS;
  int q0 = blockIdx.x * 32;

  const half_t* Q = q  + (size_t)bh * SEQ * HDIM;
  const half_t* K = k  + (size_t)bh * SEQ * HDIM;
  const half_t* V = vT + (size_t)bh * HDIM * SEQ;

  v16h aQ[2][2];
#pragma unroll
  for (int t = 0; t < 2; ++t) {
    aQ[t][0] = load_a_frag(Q, HDIM, q0 + 16 * t, 0,  lane);
    aQ[t][1] = load_a_frag(Q, HDIM, q0 + 16 * t, 32, lane);
  }

  v8f acc[2][4] = {};
  float mrun[2][8], lrun[2][8];
#pragma unroll
  for (int t = 0; t < 2; ++t)
#pragma unroll
    for (int v = 0; v < 8; ++v) { mrun[t][v] = -3.0e38f; lrun[t][v] = 0.f; }

  for (int kb = 0; kb < q0 + 32; kb += 32) {
    // K fragments loaded once, shared across both query tiles
    v16h bk[2][2];
#pragma unroll
    for (int u = 0; u < 2; ++u) {
      bk[u][0] = load_b_frag(K, HDIM, kb + 16 * u, 0,  lane);
      bk[u][1] = load_b_frag(K, HDIM, kb + 16 * u, 32, lane);
    }
    v8f sc[2][2] = {};
#pragma unroll
    for (int t = 0; t < 2; ++t)
#pragma unroll
      for (int u = 0; u < 2; ++u) {
        sc[t][u] = wmma_fp16(aQ[t][0], bk[u][0], sc[t][u]);
        sc[t][u] = wmma_fp16(aQ[t][1], bk[u][1], sc[t][u]);
      }

    int key0 = kb + ln;
    int key1 = kb + 16 + ln;
#pragma unroll
    for (int t = 0; t < 2; ++t) {
      float alpha[8];
#pragma unroll
      for (int v = 0; v < 8; ++v) {
        int qrow = q0 + 16 * t + v + hi * 8;
        float e0 = (key0 <= qrow) ? sc[t][0][v] * 0.125f : -3.0e38f;
        float e1 = (key1 <= qrow) ? sc[t][1][v] * 0.125f : -3.0e38f;
        float mx = fmaxf(e0, e1);
#pragma unroll
        for (int off = 1; off < 16; off <<= 1)
          mx = fmaxf(mx, __shfl_xor(mx, off, 32));   // stays within 16-lane half
        float mnew = fmaxf(mrun[t][v], mx);
        float al = __expf(mrun[t][v] - mnew);
        float x0 = __expf(e0 - mnew);
        float x1 = __expf(e1 - mnew);
        float rs = x0 + x1;
#pragma unroll
        for (int off = 1; off < 16; off <<= 1)
          rs += __shfl_xor(rs, off, 32);
        lrun[t][v] = lrun[t][v] * al + rs;
        mrun[t][v] = mnew;
        alpha[v] = al;
        // P (C-layout) -> LDS tile (f16)
        sP[(16 * t + v + hi * 8) * LDS_STRIDE + ln]      = (half_t)x0;
        sP[(16 * t + v + hi * 8) * LDS_STRIDE + 16 + ln] = (half_t)x1;
      }
#pragma unroll
      for (int j = 0; j < 4; ++j)
#pragma unroll
        for (int v = 0; v < 8; ++v)
          acc[t][j][v] *= alpha[v];
    }
    __syncthreads();

    v16h aP0 = load_a_frag(sP, LDS_STRIDE, 0,  0, lane);
    v16h aP1 = load_a_frag(sP, LDS_STRIDE, 16, 0, lane);
    v16h bv[4];
#pragma unroll
    for (int j = 0; j < 4; ++j)
      bv[j] = load_b_frag(V, SEQ, j * 16, kb, lane);
#pragma unroll
    for (int j = 0; j < 4; ++j) {
      acc[0][j] = wmma_fp16(aP0, bv[j], acc[0][j]);
      acc[1][j] = wmma_fp16(aP1, bv[j], acc[1][j]);
    }
    __syncthreads();
  }

#pragma unroll
  for (int t = 0; t < 2; ++t) {
    float inv[8];
#pragma unroll
    for (int v = 0; v < 8; ++v) inv[v] = 1.0f / lrun[t][v];
#pragma unroll
    for (int j = 0; j < 4; ++j)
#pragma unroll
      for (int v = 0; v < 8; ++v) {
        int m = q0 + 16 * t + v + hi * 8;
        int d = j * 16 + ln;
        ctx[((size_t)(b * SEQ + m)) * EMB + h * HDIM + d] =
            (half_t)(acc[t][j][v] * inv[v]);
      }
  }
}

// ---------------------------------------------------------------------------
// Host-side orchestration
// ---------------------------------------------------------------------------
extern "C" void kernel_launch(void* const* d_in, const int* in_sizes, int n_in,
                              void* d_out, int out_size, void* d_ws, size_t ws_size,
                              hipStream_t stream) {
  (void)in_sizes; (void)n_in; (void)out_size; (void)ws_size;

  const float* x  = (const float*)d_in[0];
  const float* Wq = (const float*)d_in[1];
  const float* Wk = (const float*)d_in[2];
  const float* Wv = (const float*)d_in[3];
  const float* Wo = (const float*)d_in[4];
  const float* bo = (const float*)d_in[5];
  const float* W1 = (const float*)d_in[6];
  const float* b1 = (const float*)d_in[7];
  const float* W2 = (const float*)d_in[8];
  const float* b2 = (const float*)d_in[9];
  const float* g1 = (const float*)d_in[10];
  const float* s1 = (const float*)d_in[11];
  const float* g2 = (const float*)d_in[12];
  const float* s2 = (const float*)d_in[13];

  size_t off = 0;
  auto alloc = [&](size_t bytes) -> char* {
    char* p = (char*)d_ws + off;
    off += (bytes + 255) & ~(size_t)255;
    return p;
  };
  half_t* WqT  = (half_t*)alloc((size_t)EMB * EMB * 2);
  half_t* WkT  = (half_t*)alloc((size_t)EMB * EMB * 2);
  half_t* WvT  = (half_t*)alloc((size_t)EMB * EMB * 2);
  half_t* WoT  = (half_t*)alloc((size_t)EMB * EMB * 2);
  half_t* W1T  = (half_t*)alloc((size_t)EMB * FFDIM * 2);
  half_t* W2T  = (half_t*)alloc((size_t)FFDIM * EMB * 2);
  half_t* h16  = (half_t*)alloc((size_t)MROWS * EMB * 2);
  half_t* q16  = (half_t*)alloc((size_t)MROWS * EMB * 2);
  half_t* k16  = (half_t*)alloc((size_t)MROWS * EMB * 2);
  half_t* v16T = (half_t*)alloc((size_t)MROWS * EMB * 2);
  half_t* c16  = (half_t*)alloc((size_t)MROWS * EMB * 2);
  float*  xres = (float*) alloc((size_t)MROWS * EMB * 4);
  half_t* h216 = (half_t*)alloc((size_t)MROWS * EMB * 2);
  half_t* u16  = (half_t*)alloc((size_t)MROWS * FFDIM * 2);

  int nEE = EMB * EMB, nEF = EMB * FFDIM;
  convert_transpose_f16<<<(nEE + 255) / 256, 256, 0, stream>>>(Wq, WqT, EMB, EMB);
  convert_transpose_f16<<<(nEE + 255) / 256, 256, 0, stream>>>(Wk, WkT, EMB, EMB);
  convert_transpose_f16<<<(nEE + 255) / 256, 256, 0, stream>>>(Wv, WvT, EMB, EMB);
  convert_transpose_f16<<<(nEE + 255) / 256, 256, 0, stream>>>(Wo, WoT, EMB, EMB);
  convert_transpose_f16<<<(nEF + 255) / 256, 256, 0, stream>>>(W1, W1T, EMB, FFDIM);
  convert_transpose_f16<<<(nEF + 255) / 256, 256, 0, stream>>>(W2, W2T, FFDIM, EMB);

  layernorm_to_f16<<<MROWS, 256, 0, stream>>>(x, g1, s1, h16);

  dim3 gEE(EMB / BN, MROWS / BM);
  gemm_wmma<<<gEE, 256, 0, stream>>>(h16, WqT, nullptr, nullptr, q16,
                                     MROWS, EMB, EMB, EPI_QK);
  gemm_wmma<<<gEE, 256, 0, stream>>>(h16, WkT, nullptr, nullptr, k16,
                                     MROWS, EMB, EMB, EPI_QK);
  gemm_wmma<<<gEE, 256, 0, stream>>>(h16, WvT, nullptr, nullptr, v16T,
                                     MROWS, EMB, EMB, EPI_VT);

  dim3 gAtt(SEQ / 32, BATCH * HEADS);
  attn_flash_wmma<<<gAtt, 32, 0, stream>>>(q16, k16, v16T, c16);

  gemm_wmma<<<gEE, 256, 0, stream>>>(c16, WoT, bo, x, xres,
                                     MROWS, EMB, EMB, EPI_RES32);

  layernorm_to_f16<<<MROWS, 256, 0, stream>>>(xres, g2, s2, h216);

  dim3 gUp(FFDIM / BN, MROWS / BM);
  gemm_wmma<<<gUp, 256, 0, stream>>>(h216, W1T, b1, nullptr, u16,
                                     MROWS, FFDIM, EMB, EPI_GELU);

  gemm_wmma<<<gEE, 256, 0, stream>>>(u16, W2T, b2, xres, (float*)d_out,
                                     MROWS, EMB, FFDIM, EPI_RES32);
}
